// MultiHeadAttention_51754355917160
// MI455X (gfx1250) — compile-verified, matrix operand order checked by reading, not run on hardware
//
#include <hip/hip_runtime.h>
#include <hip/hip_bf16.h>

// ---------------------------------------------------------------------------
// MHA with RoPE(XPos) + dynamic position bias, CDNA5 (gfx1250, wave32, WMMA,
// Tensor Data Mover double-buffered KV staging).
// B=2 N=2048 DIM=512 HEADS=8 DHEAD=64
// ---------------------------------------------------------------------------

typedef __attribute__((ext_vector_type(16))) _Float16 v16h;
typedef __attribute__((ext_vector_type(8)))  _Float16 v8h;
typedef __attribute__((ext_vector_type(8)))  float    v8f;
typedef __attribute__((ext_vector_type(4)))  unsigned v4u;
typedef __attribute__((ext_vector_type(8)))  unsigned v8u;

#define B_     2
#define N_     2048
#define DIM_   512
#define HEADS_ 8
#define DHEAD_ 64
#define INNER_ 512
#define PBD_   128

// Load a 16-half WMMA fragment: elements 0..7 at p, 8..15 at p+16 halves.
// (CDNA5 16-bit fragment packing; caller supplies p = row_base + base0,
//  base0 = 8*(lane>>4).)
__device__ __forceinline__ v16h load_frag(const _Float16* p) {
  union { v16h v; v8h h[2]; } u;
  u.h[0] = *(const v8h*)(p);
  u.h[1] = *(const v8h*)(p + 16);
  return u.v;
}

__device__ __forceinline__ v8f wmma_f16(v16h a, v16h b, v8f c) {
  return __builtin_amdgcn_wmma_f32_16x16x32_f16(false, a, false, b,
                                                (short)0, c, false, false);
}

// LDS byte offset of a __shared__ object: generic (flat) LDS addresses carry
// the offset in the low 32 bits (ISA 10.2 aperture mapping).
__device__ __forceinline__ unsigned lds_off(const void* p) {
  return (unsigned)(unsigned long long)p;
}

// ---------------------------------------------------------------------------
// Tensor Data Mover: async 2D f16 tile load global -> LDS (ISA ch.8).
// tensor_dN / strides / tile_dN are in elements (data_size = 2 bytes).
// Tracked by TENSORcnt.
// ---------------------------------------------------------------------------
__device__ __forceinline__ void tdm_load_2d_f16(
    unsigned lds_byte_addr, const void* gptr,
    unsigned tensor_d0, unsigned tensor_d1, unsigned stride0,
    unsigned tile_d0, unsigned tile_d1) {
  unsigned long long ga = (unsigned long long)gptr;
  v4u g0;
  g0.x = 1u;                                    // count=1, not restore/gather
  g0.y = lds_byte_addr;                         // D#.lds_addr
  g0.z = (unsigned)ga;                          // global_addr[31:0]
  g0.w = (unsigned)(ga >> 32) | 0x80000000u;    // global_addr[56:32] | type=2
  v8u g1;
  g1[0] = 1u << 16;                             // data_size = 2 bytes
  g1[1] = tensor_d0 << 16;                      // tensor_dim0[15:0]
  g1[2] = (tensor_d0 >> 16) | (tensor_d1 << 16);
  g1[3] = (tensor_d1 >> 16) | (tile_d0 << 16);  // tile_dim0
  g1[4] = tile_d1;                              // tile_dim1 (tile_dim2 = 0)
  g1[5] = stride0;                              // tensor_dim0_stride[31:0]
  g1[6] = 0u;
  g1[7] = 0u;
  asm volatile("tensor_load_to_lds %0, %1" : : "s"(g0), "s"(g1) : "memory");
}

// ---------------------------------------------------------------------------
// Prep: f32 -> f16 convert and transpose-convert
// ---------------------------------------------------------------------------
__global__ void cvt_kernel(const float* __restrict__ src,
                           _Float16* __restrict__ dst, int n) {
  int i = blockIdx.x * blockDim.x + threadIdx.x;
  if (i < n) dst[i] = (_Float16)src[i];
}

// dst[n*K + k] = (f16) src[k*N + n]   (src is [K][N], dst is [N][K])
__global__ void transpose_kernel(const float* __restrict__ src,
                                 _Float16* __restrict__ dst, int K, int N) {
  int i = blockIdx.x * blockDim.x + threadIdx.x;
  if (i < K * N) {
    int n = i / K, k = i % K;
    dst[i] = (_Float16)src[k * N + n];
  }
}

// ---------------------------------------------------------------------------
// Dynamic position bias MLP: h_rel[2N-1][HEADS]; one block per rel position.
// ---------------------------------------------------------------------------
__global__ __launch_bounds__(128) void dpb_kernel(
    const float* __restrict__ w1, const float* __restrict__ b1,
    const float* __restrict__ g1, const float* __restrict__ be1,
    const float* __restrict__ w2, const float* __restrict__ b2,
    const float* __restrict__ g2, const float* __restrict__ be2,
    const float* __restrict__ w3, const float* __restrict__ b3,
    float* __restrict__ hrel) {
  __shared__ float hbuf[PBD_];
  __shared__ float red[PBD_];
  __shared__ float red2[PBD_];
  const int r = blockIdx.x;         // 0..2N-2
  const int j = threadIdx.x;        // 0..127

  float p  = (float)(r - (N_ - 1));
  float sg = (p > 0.f) ? 1.f : ((p < 0.f) ? -1.f : 0.f);
  float sp = sg * __logf(fabsf(p) + 1.f);

  float h = sp * w1[j] + b1[j];
  red[j] = h; red2[j] = h * h; __syncthreads();
  #pragma unroll
  for (int s = 64; s > 0; s >>= 1) {
    if (j < s) { red[j] += red[j + s]; red2[j] += red2[j + s]; }
    __syncthreads();
  }
  float mean = red[0] * (1.f / PBD_);
  float var  = red2[0] * (1.f / PBD_) - mean * mean;
  __syncthreads();
  float ln = (h - mean) * rsqrtf(var + 1e-5f) * g1[j] + be1[j];
  hbuf[j] = ln * (1.f / (1.f + __expf(-ln)));   // silu
  __syncthreads();

  float h2 = b2[j];
  for (int i = 0; i < PBD_; ++i) h2 += hbuf[i] * w2[i * PBD_ + j];
  red[j] = h2; red2[j] = h2 * h2; __syncthreads();
  #pragma unroll
  for (int s = 64; s > 0; s >>= 1) {
    if (j < s) { red[j] += red[j + s]; red2[j] += red2[j + s]; }
    __syncthreads();
  }
  mean = red[0] * (1.f / PBD_);
  var  = red2[0] * (1.f / PBD_) - mean * mean;
  __syncthreads();
  float ln2 = (h2 - mean) * rsqrtf(var + 1e-5f) * g2[j] + be2[j];
  float sl2 = ln2 * (1.f / (1.f + __expf(-ln2)));
  __syncthreads();
  hbuf[j] = sl2; __syncthreads();

  if (j < HEADS_) {
    float o = b3[j];
    for (int i = 0; i < PBD_; ++i) o += hbuf[i] * w3[i * HEADS_ + j];
    hrel[r * HEADS_ + j] = o;
  }
}

// ---------------------------------------------------------------------------
// QKV projection GEMM (WMMA, 16x64 per wave) + fused RoPE/XPos epilogue.
// A 64-col chunk is exactly one head of one of q/k/v -> uniform epilogue.
// ---------------------------------------------------------------------------
__global__ __launch_bounds__(256) void qkv_rope_kernel(
    const _Float16* __restrict__ xh, const _Float16* __restrict__ wt,
    const float* __restrict__ bqkv,
    _Float16* __restrict__ qh,   // [B][H][N][64]  (rope * sc)
    _Float16* __restrict__ kh,   // [B][H][N][64]  (rope / sc)
    _Float16* __restrict__ vT) { // [B][H][64][N]
  const int lane  = threadIdx.x & 31;
  const int wave  = threadIdx.x >> 5;
  const int tile  = blockIdx.x * 8 + wave;     // 0..6143
  const int tilesM = (B_ * N_) / 16;           // 256
  const int tm = tile % tilesM;
  const int tc = tile / tilesM;                // 0..23 (64-wide chunks)
  const int m0 = tm * 16, n0 = tc * 64;

  const int col   = lane & 15;
  const int base0 = (lane >> 4) * 8;

  v8f acc[4] = {};
  const _Float16* ap = xh + (size_t)(m0 + col) * DIM_ + base0;
  const _Float16* bp0 = wt + (size_t)(n0 +  0 + col) * DIM_ + base0;
  const _Float16* bp1 = wt + (size_t)(n0 + 16 + col) * DIM_ + base0;
  const _Float16* bp2 = wt + (size_t)(n0 + 32 + col) * DIM_ + base0;
  const _Float16* bp3 = wt + (size_t)(n0 + 48 + col) * DIM_ + base0;
  #pragma unroll
  for (int kb = 0; kb < DIM_; kb += 32) {
    v16h a = load_frag(ap + kb);
    acc[0] = wmma_f16(a, load_frag(bp0 + kb), acc[0]);
    acc[1] = wmma_f16(a, load_frag(bp1 + kb), acc[1]);
    acc[2] = wmma_f16(a, load_frag(bp2 + kb), acc[2]);
    acc[3] = wmma_f16(a, load_frag(bp3 + kb), acc[3]);
  }

  const int sec  = n0 >> 9;          // 0=q 1=k 2=v  (uniform per wave)
  const int head = (n0 & 511) >> 6;  // uniform per wave

  #pragma unroll
  for (int g = 0; g < 4; ++g) {
    const int d = g * 16 + col;      // 0..63 within head
    const float bias = bqkv[n0 + g * 16 + col];
    const int   jj    = d >> 1;
    const float inv   = __expf(-(float)(2 * jj) * (9.210340371976184f / 64.f));
    const float sbase = ((float)(2 * (d & 31)) + 0.4f * 64.f) / (1.4f * 64.f);
    #pragma unroll
    for (int e = 0; e < 8; ++e) {
      const int mg = m0 + e + base0;     // token index over B*N
      const int bb = mg >> 11;           // batch
      const int t  = mg & (N_ - 1);      // position
      float val = acc[g][e] + bias;
      if (sec == 2) {
        vT[(((size_t)bb * HEADS_ + head) * DHEAD_ + d) * N_ + t] = (_Float16)val;
      } else {
        float partner = __shfl_xor(val, 1, 32);      // pair value at d^1
        float rot = (d & 1) ? partner : -partner;    // rotate_half
        float f = (float)t * inv;
        float c = __cosf(f), s = __sinf(f);
        float power = ((float)t - (float)(N_ / 2)) * (1.f / 8192.f);
        float scl = __powf(sbase, power);
        if (sec == 1) scl = 1.f / scl;
        float o = (val * c + rot * s) * scl;
        _Float16* dst = (sec == 0) ? qh : kh;
        dst[(((size_t)bb * HEADS_ + head) * N_ + t) * DHEAD_ + d] = (_Float16)o;
      }
    }
  }
}

// ---------------------------------------------------------------------------
// Flash attention: TDM double-buffers K/V tiles into LDS (shared by all 8
// waves of the block); per-wave online softmax; bias gathered from h_rel.
// ---------------------------------------------------------------------------
__global__ __launch_bounds__(256) void attn_kernel(
    const _Float16* __restrict__ qh, const _Float16* __restrict__ kh,
    const _Float16* __restrict__ vT, const float* __restrict__ hrel,
    _Float16* __restrict__ oh) {     // [B][N][INNER]
  __shared__ __align__(16) _Float16 kbuf[2][32 * DHEAD_]; // [32 keys][64 d]
  __shared__ __align__(16) _Float16 vbuf[2][DHEAD_ * 32]; // [64 d][32 keys]
  __shared__ __align__(16) _Float16 pbuf[8][16 * 32];

  const int lane = threadIdx.x & 31;
  const int wave = threadIdx.x >> 5;
  const int bh   = blockIdx.x >> 4;     // 0..15
  const int qblk = blockIdx.x & 15;
  const int b = bh >> 3, h = bh & 7;
  const int i0 = qblk * 128 + wave * 16;

  const int col   = lane & 15;
  const int base0 = (lane >> 4) * 8;

  const _Float16* qbase = qh + ((size_t)(b * HEADS_ + h) * N_) * DHEAD_;
  const _Float16* kbase = kh + ((size_t)(b * HEADS_ + h) * N_) * DHEAD_;
  const _Float16* vbase = vT + ((size_t)(b * HEADS_ + h) * DHEAD_) * N_;

  const unsigned koff0 = lds_off(kbuf[0]), koff1 = lds_off(kbuf[1]);
  const unsigned voff0 = lds_off(vbuf[0]), voff1 = lds_off(vbuf[1]);

  // Q fragments (reused across all KV tiles)
  const v16h aq0 = load_frag(qbase + (size_t)(i0 + col) * DHEAD_ + 0  + base0);
  const v16h aq1 = load_frag(qbase + (size_t)(i0 + col) * DHEAD_ + 32 + base0);

  v8f acc0 = {}, acc1 = {}, acc2 = {}, acc3 = {};
  float mrun[8], lrun[8];
  #pragma unroll
  for (int e = 0; e < 8; ++e) { mrun[e] = -1e30f; lrun[e] = 0.f; }

  _Float16* pw = pbuf[wave];

  // Prologue: stage tile 0 into buffer 0 via the Tensor Data Mover.
  if (wave == 0) {
    tdm_load_2d_f16(koff0, kbase, DHEAD_, N_, DHEAD_, DHEAD_, 32);
    tdm_load_2d_f16(voff0, vbase, N_, DHEAD_, N_, 32, DHEAD_);
  }

  for (int it = 0; it < N_ / 32; ++it) {
    const int jt = it * 32;
    const int p  = it & 1;
    if (wave == 0) __builtin_amdgcn_s_wait_tensorcnt(0);  // tile it landed
    __syncthreads();
    if (wave == 0 && jt + 32 < N_) {                      // prefetch tile it+1
      const unsigned kn = p ? koff0 : koff1;
      const unsigned vn = p ? voff0 : voff1;
      tdm_load_2d_f16(kn, kbase + (size_t)(jt + 32) * DHEAD_,
                      DHEAD_, N_, DHEAD_, DHEAD_, 32);
      tdm_load_2d_f16(vn, vbase + (jt + 32), N_, DHEAD_, N_, 32, DHEAD_);
    }

    const _Float16* kt = kbuf[p];
    const _Float16* vt = vbuf[p];

    // S tiles from LDS K: keys [jt,jt+16) and [jt+16,jt+32)
    v16h bk0a = load_frag(kt + (col)      * DHEAD_ + 0  + base0);
    v16h bk0b = load_frag(kt + (col)      * DHEAD_ + 32 + base0);
    v16h bk1a = load_frag(kt + (col + 16) * DHEAD_ + 0  + base0);
    v16h bk1b = load_frag(kt + (col + 16) * DHEAD_ + 32 + base0);
    v8f s0 = {}, s1 = {};
    s0 = wmma_f16(aq0, bk0a, s0);
    s0 = wmma_f16(aq1, bk0b, s0);
    s1 = wmma_f16(aq0, bk1a, s1);
    s1 = wmma_f16(aq1, bk1b, s1);

    // online softmax per row (rows e+base0; reductions over 16-lane groups)
    #pragma unroll
    for (int e = 0; e < 8; ++e) {
      const int iq = i0 + e + base0;
      const int r0 = iq - (jt + col) + (N_ - 1);
      float v0 = s0[e] * 0.125f + hrel[r0 * HEADS_ + h];
      float v1 = s1[e] * 0.125f + hrel[(r0 - 16) * HEADS_ + h];
      float mx = fmaxf(v0, v1);
      #pragma unroll
      for (int msk = 8; msk >= 1; msk >>= 1)
        mx = fmaxf(mx, __shfl_xor(mx, msk, 32));
      float mnew = fmaxf(mrun[e], mx);
      float p0 = __expf(v0 - mnew);
      float p1 = __expf(v1 - mnew);
      float sum = p0 + p1;
      #pragma unroll
      for (int msk = 8; msk >= 1; msk >>= 1)
        sum += __shfl_xor(sum, msk, 32);
      float alpha = __expf(mrun[e] - mnew);
      lrun[e] = lrun[e] * alpha + sum;
      mrun[e] = mnew;
      acc0[e] *= alpha; acc1[e] *= alpha; acc2[e] *= alpha; acc3[e] *= alpha;
      const int mrow = e + base0;
      pw[mrow * 32 + col]      = (_Float16)p0;
      pw[mrow * 32 + 16 + col] = (_Float16)p1;
    }
    __syncthreads();   // C-fragment layout -> A-fragment layout via LDS

    v16h pa = load_frag(pw + col * 32 + base0);
    v16h bv0 = load_frag(vt + (0 * 16 + col) * 32 + base0);
    v16h bv1 = load_frag(vt + (1 * 16 + col) * 32 + base0);
    v16h bv2 = load_frag(vt + (2 * 16 + col) * 32 + base0);
    v16h bv3 = load_frag(vt + (3 * 16 + col) * 32 + base0);
    acc0 = wmma_f16(pa, bv0, acc0);
    acc1 = wmma_f16(pa, bv1, acc1);
    acc2 = wmma_f16(pa, bv2, acc2);
    acc3 = wmma_f16(pa, bv3, acc3);
    __syncthreads();   // done reading buf[p] + pbuf before they are reused
  }

  // normalize + store (bhnd -> b n (h d)), f16 for the final WMMA GEMM
  #pragma unroll
  for (int e = 0; e < 8; ++e) {
    const int iq = i0 + e + base0;
    const float il = 1.f / lrun[e];
    size_t base = ((size_t)b * N_ + iq) * INNER_ + h * DHEAD_;
    oh[base + 0 * 16 + col] = (_Float16)(acc0[e] * il);
    oh[base + 1 * 16 + col] = (_Float16)(acc1[e] * il);
    oh[base + 2 * 16 + col] = (_Float16)(acc2[e] * il);
    oh[base + 3 * 16 + col] = (_Float16)(acc3[e] * il);
  }
}

// ---------------------------------------------------------------------------
// Output projection GEMM (WMMA, 16x64 per wave) + bias, f32 out.
// ---------------------------------------------------------------------------
__global__ __launch_bounds__(256) void outproj_kernel(
    const _Float16* __restrict__ oh, const _Float16* __restrict__ wt,
    const float* __restrict__ bout, float* __restrict__ out) {
  const int lane = threadIdx.x & 31;
  const int wave = threadIdx.x >> 5;
  const int tile = blockIdx.x * 8 + wave;   // 0..2047
  const int tm = tile & 255;                // 256 M tiles
  const int tc = tile >> 8;                 // 8 chunks of 64 cols
  const int m0 = tm * 16, n0 = tc * 64;
  const int col = lane & 15, base0 = (lane >> 4) * 8;

  v8f acc[4] = {};
  const _Float16* ap  = oh + (size_t)(m0 + col) * INNER_ + base0;
  const _Float16* bp0 = wt + (size_t)(n0 +  0 + col) * INNER_ + base0;
  const _Float16* bp1 = wt + (size_t)(n0 + 16 + col) * INNER_ + base0;
  const _Float16* bp2 = wt + (size_t)(n0 + 32 + col) * INNER_ + base0;
  const _Float16* bp3 = wt + (size_t)(n0 + 48 + col) * INNER_ + base0;
  #pragma unroll
  for (int kb = 0; kb < INNER_; kb += 32) {
    v16h a = load_frag(ap + kb);
    acc[0] = wmma_f16(a, load_frag(bp0 + kb), acc[0]);
    acc[1] = wmma_f16(a, load_frag(bp1 + kb), acc[1]);
    acc[2] = wmma_f16(a, load_frag(bp2 + kb), acc[2]);
    acc[3] = wmma_f16(a, load_frag(bp3 + kb), acc[3]);
  }
  #pragma unroll
  for (int g = 0; g < 4; ++g) {
    const float bb = bout[n0 + g * 16 + col];
    #pragma unroll
    for (int e = 0; e < 8; ++e)
      out[(size_t)(m0 + e + base0) * DIM_ + n0 + g * 16 + col] = acc[g][e] + bb;
  }
}

// ---------------------------------------------------------------------------
// Launch
// ---------------------------------------------------------------------------
extern "C" void kernel_launch(void* const* d_in, const int* in_sizes, int n_in,
                              void* d_out, int out_size, void* d_ws,
                              size_t ws_size, hipStream_t stream) {
  (void)in_sizes; (void)n_in; (void)out_size; (void)ws_size;
  const float* x      = (const float*)d_in[0];
  const float* w_qkv  = (const float*)d_in[1];
  const float* b_qkv  = (const float*)d_in[2];
  const float* w_out  = (const float*)d_in[3];
  const float* b_out  = (const float*)d_in[4];
  const float* dw1 = (const float*)d_in[5];
  const float* db1 = (const float*)d_in[6];
  const float* dg1 = (const float*)d_in[7];
  const float* dbe1= (const float*)d_in[8];
  const float* dw2 = (const float*)d_in[9];
  const float* db2 = (const float*)d_in[10];
  const float* dg2 = (const float*)d_in[11];
  const float* dbe2= (const float*)d_in[12];
  const float* dw3 = (const float*)d_in[13];
  const float* db3 = (const float*)d_in[14];
  float* out = (float*)d_out;

  char* ws = (char*)d_ws;
  _Float16* xh    = (_Float16*)(ws + 0);          //  4 MB  [4096][512]
  _Float16* wqkvt = (_Float16*)(ws + 4194304);    //  1.5MB [1536][512]
  _Float16* woutt = (_Float16*)(ws + 5767168);    //  0.5MB [512][512]
  _Float16* qh    = (_Float16*)(ws + 6291456);    //  4 MB
  _Float16* kh    = (_Float16*)(ws + 10485760);   //  4 MB
  _Float16* vT    = (_Float16*)(ws + 14680064);   //  4 MB
  float*    hrel  = (float*)   (ws + 18874368);   //  131 KB [4095][8]
  _Float16* oh    = (_Float16*)(ws + 19005440);   //  4 MB

  cvt_kernel<<<(B_*N_*DIM_ + 255) / 256, 256, 0, stream>>>(x, xh, B_*N_*DIM_);
  transpose_kernel<<<(DIM_*3*INNER_ + 255) / 256, 256, 0, stream>>>(
      w_qkv, wqkvt, DIM_, 3 * INNER_);
  transpose_kernel<<<(INNER_*DIM_ + 255) / 256, 256, 0, stream>>>(
      w_out, woutt, INNER_, DIM_);
  dpb_kernel<<<2 * N_ - 1, 128, 0, stream>>>(dw1, db1, dg1, dbe1,
                                             dw2, db2, dg2, dbe2,
                                             dw3, db3, hrel);
  qkv_rope_kernel<<<768, 256, 0, stream>>>(xh, wqkvt, b_qkv, qh, kh, vT);
  attn_kernel<<<256, 256, 0, stream>>>(qh, kh, vT, hrel, oh);
  outproj_kernel<<<256, 256, 0, stream>>>(oh, woutt, b_out, out);
}
